// MultiHeadAttention_9835475108528
// MI455X (gfx1250) — matrix operand order repaired and checked by reading, not compiled, not run on hardware
//
#include <hip/hip_runtime.h>
#include <cstdint>
#include <cstddef>

// ---------------------------------------------------------------------------
// MultiHeadAttention for MI455X (gfx1250): bf16 WMMA everywhere, flash-style
// attention with online softmax, V stored transposed for contiguous PV frags,
// K/V tiles staged through LDS via async global->LDS copies (double-buffered).
// ---------------------------------------------------------------------------

typedef __bf16 bf16;
typedef __attribute__((ext_vector_type(8)))  bf16  v8bf;
typedef __attribute__((ext_vector_type(16))) bf16  v16bf;
typedef __attribute__((ext_vector_type(8)))  float v8f;
typedef __attribute__((ext_vector_type(4)))  int   v4i;

#define DIM   1024
#define SEQ   2048
#define NH    16
#define HD    64
#define BATCH 4
#define MROWS (BATCH * SEQ)   // 8192 rows for all projection GEMMs

#define AS1 __attribute__((address_space(1)))
#define AS3 __attribute__((address_space(3)))

#if __has_builtin(__builtin_amdgcn_global_load_async_to_lds_b128)
#define HAVE_ASYNC_LDS 1
#endif

__device__ __forceinline__ bf16 to_bf16(float f) {
  uint32_t u = __builtin_bit_cast(uint32_t, f);
  uint32_t r = u + 0x7FFFu + ((u >> 16) & 1u);   // round-to-nearest-even
  uint16_t h = (uint16_t)(r >> 16);
  return __builtin_bit_cast(bf16, h);
}

__device__ __forceinline__ v16bf cat8(v8bf lo, v8bf hi) {
  return __builtin_shufflevector(lo, hi, 0,1,2,3,4,5,6,7,8,9,10,11,12,13,14,15);
}

// 16-byte global -> LDS copy: async (ASYNCcnt) when available, else via VGPRs.
__device__ __forceinline__ void copy16_g2l(const bf16* g, bf16* l) {
#if defined(HAVE_ASYNC_LDS)
  __builtin_amdgcn_global_load_async_to_lds_b128((AS1 v4i*)g, (AS3 v4i*)l, 0, 0);
#else
  *(v8bf*)l = *(const v8bf*)g;
#endif
}

// -------------------------------- f32 -> bf16 ------------------------------
__global__ __launch_bounds__(256)
void cvt_f32_bf16(const float* __restrict__ in, bf16* __restrict__ out, int n) {
  int idx = (blockIdx.x * blockDim.x + threadIdx.x) * 4;
  if (idx + 3 < n) {
    float4 f = *(const float4*)(in + idx);
    out[idx + 0] = to_bf16(f.x);
    out[idx + 1] = to_bf16(f.y);
    out[idx + 2] = to_bf16(f.z);
    out[idx + 3] = to_bf16(f.w);
  }
}

// ------------------------- projection GEMM: Y = A @ W^T --------------------
// A: (MROWS x 1024) bf16 row-major, W: (1024 x 1024) bf16 row-major.
// Block = 128x128 tile, 256 threads = 8 waves, wave tile 64x32 (4x2 WMMA).
// mode 0: head-split store [B,H,S,64] bf16   (Q with scale, K)
// mode 1: transposed head-split store [B,H,64,S] bf16   (V^T)
// mode 2: plain f32 store (M x 1024)   (final output)
__global__ __launch_bounds__(256)
void gemm_proj(const bf16* __restrict__ A, const bf16* __restrict__ W,
               const float* __restrict__ bias, void* __restrict__ out,
               int mode, float scale) {
  const int lane = threadIdx.x & 31;
  const int wave = threadIdx.x >> 5;
  const int half = lane >> 4;
  const int ln   = lane & 15;
  const int wm   = wave & 1;        // 2 waves along M
  const int wn   = wave >> 1;       // 4 waves along N
  const int m0   = blockIdx.y * 128 + wm * 64;
  const int n0   = blockIdx.x * 128 + wn * 32;

  v8f acc[4][2];
  #pragma unroll
  for (int mt = 0; mt < 4; ++mt)
    #pragma unroll
    for (int nt = 0; nt < 2; ++nt)
      #pragma unroll
      for (int r = 0; r < 8; ++r) acc[mt][nt][r] = 0.0f;

  for (int k0 = 0; k0 < DIM; k0 += 32) {
    v16bf afrag[4];
    #pragma unroll
    for (int mt = 0; mt < 4; ++mt) {
      const bf16* p = A + (size_t)(m0 + mt * 16 + ln) * DIM + k0 + half * 8;
      v8bf lo = *(const v8bf*)(p);
      v8bf hi = *(const v8bf*)(p + 16);
      afrag[mt] = cat8(lo, hi);
    }
    v16bf bfrag[2];
    #pragma unroll
    for (int nt = 0; nt < 2; ++nt) {
      const bf16* p = W + (size_t)(n0 + nt * 16 + ln) * DIM + k0 + half * 8;
      bfrag[nt] = cat8(*(const v8bf*)(p), *(const v8bf*)(p + 16));
    }
    #pragma unroll
    for (int mt = 0; mt < 4; ++mt)
      #pragma unroll
      for (int nt = 0; nt < 2; ++nt)
        acc[mt][nt] = __builtin_amdgcn_wmma_f32_16x16x32_bf16(
            false, afrag[mt], false, bfrag[nt], (short)0, acc[mt][nt],
            false, false);
  }

  // Epilogue: bias + scale, then store per mode.
  #pragma unroll
  for (int nt = 0; nt < 2; ++nt) {
    const int n  = n0 + nt * 16 + ln;
    const float bs = bias[n];
    #pragma unroll
    for (int mt = 0; mt < 4; ++mt) {
      #pragma unroll
      for (int r = 0; r < 8; ++r) {
        const int m = m0 + mt * 16 + half * 8 + r;
        const float v = (acc[mt][nt][r] + bs) * scale;
        if (mode == 2) {
          ((float*)out)[(size_t)m * DIM + n] = v;
        } else {
          const int b = m >> 11, s = m & (SEQ - 1);
          const int h = n >> 6, d = n & (HD - 1);
          if (mode == 0)        // [B,H,S,64]
            ((bf16*)out)[(((size_t)(b * NH + h)) * SEQ + s) * HD + d] = to_bf16(v);
          else                  // [B,H,64,S]  (V transposed)
            ((bf16*)out)[(((size_t)(b * NH + h)) * HD + d) * SEQ + s] = to_bf16(v);
        }
      }
    }
  }
}

// ------------------------------ flash attention ----------------------------
// Q: [B,H,S,64] bf16, K: [B,H,S,64] bf16, Vt: [B,H,64,S] bf16.
// Grid (S/128, H, B), 256 threads = 8 waves; each wave owns 16 q-rows.
// K/V 64-step tiles staged once per block in LDS via async copies
// (double-buffered, prefetch next tile while computing current).
__global__ __launch_bounds__(256)
void flash_attn(const bf16* __restrict__ Q, const bf16* __restrict__ K,
                const bf16* __restrict__ Vt, const unsigned char* __restrict__ mask,
                bf16* __restrict__ out) {
  __shared__ bf16 ldsK[2][64 * 64];   // 2 x 8 KB  : K tile, rows = key pos
  __shared__ bf16 ldsV[2][64 * 64];   // 2 x 8 KB  : V^T tile, rows = head dim
  __shared__ bf16 ldsP[8][16 * 64];   // 16 KB     : per-wave P staging

  const int tid  = threadIdx.x;
  const int lane = tid & 31;
  const int wave = tid >> 5;
  const int half = lane >> 4;
  const int ln   = lane & 15;
  const int b = blockIdx.z, h = blockIdx.y;
  const int q0 = blockIdx.x * 128 + wave * 16;

  const bf16* Qbh = Q  + ((size_t)(b * NH + h)) * SEQ * HD;
  const bf16* Kbh = K  + ((size_t)(b * NH + h)) * SEQ * HD;
  const bf16* Vbh = Vt + ((size_t)(b * NH + h)) * HD * SEQ;
  const unsigned char* mk = mask + (size_t)b * SEQ;

  // Issue one 64-key tile of K and V into LDS buffer `buf`.
  // 512 chunks of 16B per matrix; 256 threads x 2 chunks each.
  auto issue_tile = [&](int kbase, int buf) {
    #pragma unroll
    for (int c = 0; c < 2; ++c) {
      const int chunk = tid + c * 256;          // 0..511
      const int row = chunk >> 3;               // 0..63
      const int col = (chunk & 7) * 8;          // bf16 offset, 16B chunks
      copy16_g2l(Kbh + (size_t)(kbase + row) * HD + col,
                 &ldsK[buf][row * 64 + col]);
      copy16_g2l(Vbh + (size_t)row * SEQ + kbase + col,
                 &ldsV[buf][row * 64 + col]);
    }
  };

  // Q A-fragments for this wave's 16 rows (reused for every K step).
  v16bf aq[2];
  #pragma unroll
  for (int i = 0; i < 2; ++i) {
    const bf16* p = Qbh + (size_t)(q0 + ln) * HD + i * 32 + half * 8;
    aq[i] = cat8(*(const v8bf*)(p), *(const v8bf*)(p + 16));
  }

  v8f oacc[4];
  #pragma unroll
  for (int t = 0; t < 4; ++t)
    #pragma unroll
    for (int r = 0; r < 8; ++r) oacc[t][r] = 0.0f;

  float rowmax[8], rowsum[8];
  #pragma unroll
  for (int r = 0; r < 8; ++r) { rowmax[r] = -__builtin_inff(); rowsum[r] = 0.0f; }

  bf16* lp = &ldsP[wave][0];

  issue_tile(0, 0);                              // prologue: fill buffer 0

  const int nsteps = SEQ / 64;                   // 32
  for (int it = 0; it < nsteps; ++it) {
    const int k0  = it * 64;
    const int cur = it & 1;
    // Prefetch next tile into the other buffer (wrap keeps it in-bounds;
    // the trailing prefetch is drained by S_ENDPGM's implicit wait-idle).
    issue_tile((k0 + 64) & (SEQ - 1), cur ^ 1);
#if defined(HAVE_ASYNC_LDS)
    // 4 newest outstanding asyncs = the prefetch; <=4 drains current tile.
    asm volatile("s_wait_asynccnt 4" ::: "memory");
#endif
    __syncthreads();                             // current tile visible to all

    // ---- S = Q @ K^T : 16x64 logits in 4 accumulators (B-frags from LDS) --
    v8f sacc[4];
    #pragma unroll
    for (int t = 0; t < 4; ++t)
      #pragma unroll
      for (int r = 0; r < 8; ++r) sacc[t][r] = 0.0f;

    #pragma unroll
    for (int t = 0; t < 4; ++t) {
      #pragma unroll
      for (int i = 0; i < 2; ++i) {
        const bf16* kp = &ldsK[cur][(t * 16 + ln) * 64 + i * 32 + half * 8];
        v16bf bk = cat8(*(const v8bf*)(kp), *(const v8bf*)(kp + 16));
        sacc[t] = __builtin_amdgcn_wmma_f32_16x16x32_bf16(
            false, aq[i], false, bk, (short)0, sacc[t], false, false);
      }
    }

    // ---- key-position mask (lane's column across its 8 rows) ----
    #pragma unroll
    for (int t = 0; t < 4; ++t) {
      if (mk[k0 + t * 16 + ln]) {
        #pragma unroll
        for (int r = 0; r < 8; ++r) sacc[t][r] = -__builtin_inff();
      }
    }

    // ---- online softmax: per-row max over 64 cols (16-lane halves) ----
    float mx[8];
    #pragma unroll
    for (int r = 0; r < 8; ++r) {
      float m = sacc[0][r];
      #pragma unroll
      for (int t = 1; t < 4; ++t) m = fmaxf(m, sacc[t][r]);
      m = fmaxf(m, __shfl_xor(m, 1, 32));
      m = fmaxf(m, __shfl_xor(m, 2, 32));
      m = fmaxf(m, __shfl_xor(m, 4, 32));
      m = fmaxf(m, __shfl_xor(m, 8, 32));
      mx[r] = m;
    }

    float corr[8];
    #pragma unroll
    for (int r = 0; r < 8; ++r) {
      const float mnew = fmaxf(rowmax[r], mx[r]);
      corr[r] = __expf(rowmax[r] - mnew);
      rowmax[r] = mnew;
    }

    // ---- P = exp(S - max), partial row sums ----
    float rs[8];
    #pragma unroll
    for (int r = 0; r < 8; ++r) rs[r] = 0.0f;
    #pragma unroll
    for (int t = 0; t < 4; ++t)
      #pragma unroll
      for (int r = 0; r < 8; ++r) {
        const float p = __expf(sacc[t][r] - rowmax[r]);
        sacc[t][r] = p;
        rs[r] += p;
      }
    #pragma unroll
    for (int r = 0; r < 8; ++r) {
      float s = rs[r];
      s += __shfl_xor(s, 1, 32);
      s += __shfl_xor(s, 2, 32);
      s += __shfl_xor(s, 4, 32);
      s += __shfl_xor(s, 8, 32);
      rowsum[r] = rowsum[r] * corr[r] + s;
    }

    // ---- rescale O by correction ----
    #pragma unroll
    for (int t = 0; t < 4; ++t)
      #pragma unroll
      for (int r = 0; r < 8; ++r) oacc[t][r] *= corr[r];

    // ---- stage P through LDS: accumulator layout -> A-fragment layout ----
    #pragma unroll
    for (int t = 0; t < 4; ++t)
      #pragma unroll
      for (int r = 0; r < 8; ++r)
        lp[(half * 8 + r) * 64 + t * 16 + ln] = to_bf16(sacc[t][r]);

    asm volatile("s_wait_dscnt 0" ::: "memory");   // wave-local LDS RAW fence

    v16bf ap[2];
    #pragma unroll
    for (int i = 0; i < 2; ++i) {
      const bf16* pp = lp + ln * 64 + i * 32 + half * 8;
      ap[i] = cat8(*(const v8bf*)(pp), *(const v8bf*)(pp + 16));
    }

    // ---- O += P @ V  (B-frags from LDS V^T tile) ----
    #pragma unroll
    for (int t = 0; t < 4; ++t) {
      #pragma unroll
      for (int i = 0; i < 2; ++i) {
        const bf16* vp = &ldsV[cur][(t * 16 + ln) * 64 + i * 32 + half * 8];
        v16bf bv = cat8(*(const v8bf*)(vp), *(const v8bf*)(vp + 16));
        oacc[t] = __builtin_amdgcn_wmma_f32_16x16x32_bf16(
            false, ap[i], false, bv, (short)0, oacc[t], false, false);
      }
    }

    __syncthreads();   // all waves done reading before next tile overwrite
  }

  // ---- normalize and store merged-head output [B,S,DIM] bf16 ----
  float inv[8];
  #pragma unroll
  for (int r = 0; r < 8; ++r) inv[r] = 1.0f / rowsum[r];
  #pragma unroll
  for (int t = 0; t < 4; ++t) {
    const int col = h * HD + t * 16 + ln;
    #pragma unroll
    for (int r = 0; r < 8; ++r) {
      const int s = q0 + half * 8 + r;
      out[((size_t)(b * SEQ + s)) * DIM + col] = to_bf16(oacc[t][r] * inv[r]);
    }
  }
}

// --------------------------------- launcher --------------------------------
extern "C" void kernel_launch(void* const* d_in, const int* in_sizes, int n_in,
                              void* d_out, int out_size, void* d_ws, size_t ws_size,
                              hipStream_t stream) {
  (void)in_sizes; (void)n_in; (void)out_size; (void)ws_size;

  const float* x    = (const float*)d_in[0];
  const float* mem  = (const float*)d_in[1];
  const unsigned char* mask = (const unsigned char*)d_in[2];
  const float* wq = (const float*)d_in[3];
  const float* bq = (const float*)d_in[4];
  const float* wk = (const float*)d_in[5];
  const float* bk = (const float*)d_in[6];
  const float* wv = (const float*)d_in[7];
  const float* bv = (const float*)d_in[8];
  const float* wo = (const float*)d_in[9];
  const float* bo = (const float*)d_in[10];
  float* out = (float*)d_out;

  char* ws = (char*)d_ws;
  size_t off = 0;
  auto salloc = [&](size_t bytes) -> char* {
    char* p = ws + off;
    off += (bytes + 255) & ~(size_t)255;
    return p;
  };
  const size_t actBytes = (size_t)MROWS * DIM * sizeof(bf16);   // 16 MB each
  const size_t wBytes   = (size_t)DIM * DIM * sizeof(bf16);     // 2 MB each
  bf16* xb  = (bf16*)salloc(actBytes);
  bf16* mb  = (bf16*)salloc(actBytes);
  bf16* wqb = (bf16*)salloc(wBytes);
  bf16* wkb = (bf16*)salloc(wBytes);
  bf16* wvb = (bf16*)salloc(wBytes);
  bf16* wob = (bf16*)salloc(wBytes);
  bf16* Qb  = (bf16*)salloc(actBytes);   // [B,H,S,64]
  bf16* Kb  = (bf16*)salloc(actBytes);   // [B,H,S,64]
  bf16* Vtb = (bf16*)salloc(actBytes);   // [B,H,64,S]
  bf16* AOb = (bf16*)salloc(actBytes);   // [B,S,DIM] merged heads

  const int nAct = MROWS * DIM;          // 8,388,608
  const int nW   = DIM * DIM;            // 1,048,576
  const int cvtBlkA = (nAct / 4 + 255) / 256;
  const int cvtBlkW = (nW / 4 + 255) / 256;
  cvt_f32_bf16<<<cvtBlkA, 256, 0, stream>>>(x,   xb, nAct);
  cvt_f32_bf16<<<cvtBlkA, 256, 0, stream>>>(mem, mb, nAct);
  cvt_f32_bf16<<<cvtBlkW, 256, 0, stream>>>(wq, wqb, nW);
  cvt_f32_bf16<<<cvtBlkW, 256, 0, stream>>>(wk, wkb, nW);
  cvt_f32_bf16<<<cvtBlkW, 256, 0, stream>>>(wv, wvb, nW);
  cvt_f32_bf16<<<cvtBlkW, 256, 0, stream>>>(wo, wob, nW);

  dim3 gg(DIM / 128, MROWS / 128);       // (8, 64)
  const float qscale = 0.125f;           // 1/sqrt(64)
  gemm_proj<<<gg, 256, 0, stream>>>(xb, wqb, bq, Qb,  0, qscale);
  gemm_proj<<<gg, 256, 0, stream>>>(mb, wkb, bk, Kb,  0, 1.0f);
  gemm_proj<<<gg, 256, 0, stream>>>(mb, wvb, bv, Vtb, 1, 1.0f);

  flash_attn<<<dim3(SEQ / 128, NH, BATCH), 256, 0, stream>>>(Qb, Kb, Vtb, mask, AOb);

  gemm_proj<<<gg, 256, 0, stream>>>(AOb, wob, bo, out, 2, 1.0f);
}
